// SpinalCordDraft_47347719471599
// MI455X (gfx1250) — matrix operand order repaired and checked by reading, not compiled, not run on hardware
//
#include <hip/hip_runtime.h>

// ---------------------------------------------------------------------------
// Model constants (from the reference)
// ---------------------------------------------------------------------------
static constexpr int cV = 32000;
static constexpr int cD = 1024;
static constexpr int cH = 16;
static constexpr int cHKV = 4;
static constexpr int cFF = 4096;
static constexpr int cL = 8;
static constexpr int cHD = 64;          // D / H
static constexpr int cB = 2;
static constexpr int cS = 1024;
static constexpr int cM = cB * cS;      // 2048 token rows

typedef __bf16 bf16;
typedef bf16  v4bf  __attribute__((ext_vector_type(4)));
typedef bf16  v8bf  __attribute__((ext_vector_type(8)));
typedef bf16  v16bf __attribute__((ext_vector_type(16)));
typedef float v8f   __attribute__((ext_vector_type(8)));
typedef unsigned int u32x4 __attribute__((ext_vector_type(4)));
typedef int          i32x4 __attribute__((ext_vector_type(4)));
typedef int          i32x8 __attribute__((ext_vector_type(8)));

__device__ __forceinline__ v16bf cat8(v8bf lo, v8bf hi) {
    return __builtin_shufflevector(lo, hi, 0, 1, 2, 3, 4, 5, 6, 7,
                                           8, 9, 10, 11, 12, 13, 14, 15);
}

__device__ __forceinline__ v8f wmma_bf16(v16bf a, v16bf b, v8f c) {
    // D(f32 16x16) = A(bf16 16x32) * B(bf16 32x16) + C
    return __builtin_amdgcn_wmma_f32_16x16x32_bf16(
        /*neg_a=*/false, a, /*neg_b=*/false, b,
        /*c_mod=*/(short)0, c, /*reuse_a=*/false, /*reuse_b=*/false);
}

// ---------------------------------------------------------------------------
// 1. Embedding gather:  x[m,:] = embed[tokens[m],:]   (fp32)
// ---------------------------------------------------------------------------
__global__ __launch_bounds__(256) void embed_kernel(const int* __restrict__ tokens,
                                                    const float* __restrict__ emb,
                                                    float* __restrict__ x) {
    int row = blockIdx.x;
    int tid = threadIdx.x;
    int tok = tokens[row];
    float4 v = *(const float4*)(emb + (size_t)tok * cD + tid * 4);
    *(float4*)(x + (size_t)row * cD + tid * 4) = v;
}

// ---------------------------------------------------------------------------
// 2. RMSNorm -> bf16:  out[m,:] = bf16(x * rsqrt(mean(x^2)+eps) * w)
// ---------------------------------------------------------------------------
__global__ __launch_bounds__(256) void rmsnorm_kernel(const float* __restrict__ x,
                                                      const float* __restrict__ w,
                                                      bf16* __restrict__ out) {
    __shared__ float red[8];
    __shared__ float sscale;
    int row = blockIdx.x;
    int tid = threadIdx.x;
    const float* xr = x + (size_t)row * cD;
    float4 v = *(const float4*)(xr + tid * 4);
    float ss = v.x * v.x + v.y * v.y + v.z * v.z + v.w * v.w;
#pragma unroll
    for (int m = 16; m >= 1; m >>= 1) ss += __shfl_xor(ss, m);
    if ((tid & 31) == 0) red[tid >> 5] = ss;
    __syncthreads();
    if (tid == 0) {
        float tot = 0.f;
#pragma unroll
        for (int i = 0; i < 8; ++i) tot += red[i];
        sscale = rsqrtf(tot * (1.0f / cD) + 1e-6f);
    }
    __syncthreads();
    float sc = sscale;
    float4 wv = *(const float4*)(w + tid * 4);
    v4bf o = { (bf16)(v.x * sc * wv.x), (bf16)(v.y * sc * wv.y),
               (bf16)(v.z * sc * wv.z), (bf16)(v.w * sc * wv.w) };
    *(v4bf*)(out + (size_t)row * cD + tid * 4) = o;
}

// ---------------------------------------------------------------------------
// 3. Tiled bf16 WMMA GEMM with TDM double buffering.
//    C[M,N](f32) = A[M,K](bf16) @ B(f32->bf16) (+resid)
//    BT=false: B row-major [K,N].  BT=true: B row-major [N,K] (B^T).
//    Tile: BM=64, BN=128, KB=32. Block = 256 threads = 8 waves (4x2),
//    each wave computes a 16x64 strip = 4 WMMA accumulators.
//    A-tile (pure bf16 copy) is staged by the Tensor Data Mover
//    (tensor_load_to_lds, TENSORcnt); B-tile needs fp32->bf16 conversion so it
//    uses VALU loads + transposed LDS stores. One barrier per k-step:
//    tile kt+1 is prefetched while tile kt feeds the WMMAs.
// ---------------------------------------------------------------------------
template <bool BT, bool RES>
__global__ __launch_bounds__(256) void gemm_kernel(const bf16* __restrict__ A,
                                                   const float* __restrict__ B,
                                                   float* __restrict__ C,
                                                   const float* __restrict__ resid,
                                                   int M, int N, int K) {
    __shared__ bf16 sA[2][64 * 32];      // [row][k]
    __shared__ bf16 sB[2][128 * 32];     // [n][k]  (transposed for contiguous frags)

    int tid  = threadIdx.x;
    int lane = tid & 31;
    int wv   = tid >> 5;
    int mblk = blockIdx.y * 64;
    int nblk = blockIdx.x * 128;
    int mb   = (wv & 3) * 16;
    int nb   = (wv >> 2) * 64;
    int lrow = lane & 15;
    int half = lane >> 4;                // 0: lanes 0-15, 1: lanes 16-31

    v8f zero8 = { 0.f, 0.f, 0.f, 0.f, 0.f, 0.f, 0.f, 0.f };
    v8f acc[4];
#pragma unroll
    for (int i = 0; i < 4; ++i) acc[i] = zero8;

    int nk = K >> 5;

    // ---- TDM: DMA the 64x32(bf16) A tile (64 rows x 64B, row stride K*2 B)
    //      into LDS. Issued by wave 0 only; all descriptor operands uniform.
    //      D# per CDNA5 ISA 8.3/8.4: 2-D tensor, element size 8B.
    auto tdm_loadA = [&](int kt, int buf) {
        unsigned long long ga =
            (unsigned long long)(size_t)(A + (size_t)mblk * K + kt * 32);
        unsigned ldsoff = (unsigned)(size_t)(&sA[buf][0]);  // low 32b = LDS byte offset
        unsigned k8   = (unsigned)(K >> 2);                 // row length/stride in 8B units
        unsigned rows = (unsigned)M;
        u32x4 g0 = { 1u,                                    // count = 1 valid descriptor
                     ldsoff,                                // lds_addr
                     (unsigned)ga,                          // global_addr[31:0]
                     (unsigned)(ga >> 32) | (2u << 30) };   // global_addr[56:32] | type=2
        i32x8 g1;
        g1[0] = (int)(3u << 16);                            // data_size = 8 bytes
        g1[1] = (int)((k8 & 0xFFFFu) << 16);                // tensor_dim0[15:0]
        g1[2] = (int)(((k8 >> 16) & 0xFFFFu) |
                      ((rows & 0xFFFFu) << 16));            // td0[31:16] | td1[15:0]
        g1[3] = (int)(((rows >> 16) & 0xFFFFu) |
                      (8u << 16));                          // td1[31:16] | tile_dim0 = 8
        g1[4] = 64;                                         // tile_dim1 = 64, tile_dim2 = 0
        g1[5] = (int)k8;                                    // tensor_dim0_stride[31:0]
        g1[6] = 0;                                          // stride hi / dim1_stride
        g1[7] = 0;
        i32x4 gz  = { 0, 0, 0, 0 };                         // groups 2/3 unused (2-D)
        i32x8 gz8 = { 0, 0, 0, 0, 0, 0, 0, 0 };            // trailing group (clang-23 form)
        __builtin_amdgcn_tensor_load_to_lds(g0, g1, gz, gz, gz8, 0);
    };

    // ---- B tile: fp32 -> bf16 conversion, stored transposed [n][k]
    auto loadB = [&](int kt, int buf) {
        int k0  = kt * 32;
        int n   = tid & 127;
        int ksb = (tid >> 7) * 16;
        if (BT) {
#pragma unroll
            for (int i = 0; i < 4; ++i) {
                int kk = ksb + i * 4;
                float4 f = *(const float4*)(B + (size_t)(nblk + n) * K + k0 + kk);
                v4bf p = { (bf16)f.x, (bf16)f.y, (bf16)f.z, (bf16)f.w };
                *(v4bf*)&sB[buf][n * 32 + kk] = p;
            }
        } else {
            size_t Ns = (size_t)N;
#pragma unroll
            for (int i = 0; i < 4; ++i) {
                int kk = ksb + i * 4;
                const float* bp = B + (size_t)(k0 + kk) * Ns + (nblk + n);
                v4bf p = { (bf16)bp[0], (bf16)bp[Ns], (bf16)bp[2 * Ns], (bf16)bp[3 * Ns] };
                *(v4bf*)&sB[buf][n * 32 + kk] = p;
            }
        }
    };

    // ---- prologue: stage tile 0
    if (wv == 0) tdm_loadA(0, 0);
    loadB(0, 0);

    for (int kt = 0; kt < nk; ++kt) {
        int buf = kt & 1;
        // Tile kt is the only outstanding TDM op of wave 0 here.
        if (wv == 0) __builtin_amdgcn_s_wait_tensorcnt(0);
        // Publish tile kt to all waves; also retires all reads of tile kt-1,
        // so buffer buf^1 is free to overwrite below.
        __syncthreads();
        if (kt + 1 < nk) {
            if (wv == 0) tdm_loadA(kt + 1, buf ^ 1);      // DMA next A tile
            loadB(kt + 1, buf ^ 1);                       // prefetch next B tile
            if (kt + 2 < nk) {                            // warm L2 for tile kt+2
                size_t poff = BT ? ((size_t)(nblk + (tid & 127)) * K + (kt + 2) * 32)
                                 : ((size_t)((kt + 2) * 32 + (tid >> 7) * 16) * N
                                    + nblk + (tid & 127));
                __builtin_prefetch(B + poff, 0, 1);
            }
        }
        // ---- A fragment (16x32 bf16): lanes<16 K{0..7,16..23}, lanes>=16 K{8..15,24..31}
        const bf16* ar = &sA[buf][(mb + lrow) * 32];
        v16bf afrag = cat8(*(const v8bf*)(ar + half * 8),
                           *(const v8bf*)(ar + 16 + half * 8));
        // ---- B fragments + WMMA
#pragma unroll
        for (int nf = 0; nf < 4; ++nf) {
            const bf16* br = &sB[buf][(nb + nf * 16 + lrow) * 32 + half * 8];
            v16bf bfrag = cat8(*(const v8bf*)br, *(const v8bf*)(br + 16));
            acc[nf] = wmma_bf16(afrag, bfrag, acc[nf]);
        }
    }

    // ---- epilogue: C layout = lanes<16 rows 0..7, lanes>=16 rows 8..15
#pragma unroll
    for (int nf = 0; nf < 4; ++nf) {
#pragma unroll
        for (int r = 0; r < 8; ++r) {
            int row = mblk + mb + (half ? r + 8 : r);
            int col = nblk + nb + nf * 16 + lrow;
            size_t idx = (size_t)row * N + col;
            float v = acc[nf][r];
            if (RES) v += resid[idx];
            C[idx] = v;
        }
    }
}

// ---------------------------------------------------------------------------
// 4a. RoPE + repack: src fp32 [B*S, nh*64]  ->  dst bf16 [B, nh, S, 64]
// ---------------------------------------------------------------------------
__global__ __launch_bounds__(256) void rope_pack_kernel(const float* __restrict__ src,
                                                        bf16* __restrict__ dst,
                                                        int nh, int do_rope) {
    int t = blockIdx.x * blockDim.x + threadIdx.x;
    int total = cB * nh * cS * 32;
    if (t >= total) return;
    int d    = t & 31;
    int s    = (t >> 5) & (cS - 1);
    int rest = t >> 15;
    int hh   = rest % nh;
    int b    = rest / nh;
    const float* sp = src + ((size_t)(b * cS + s) * nh + hh) * 64;
    float x1 = sp[d], x2 = sp[d + 32];
    float o1 = x1, o2 = x2;
    if (do_rope) {
        float inv = __powf(10000.0f, -(float)d * (1.0f / 32.0f));
        float ang = (float)s * inv;
        float sn, cc;
        __sincosf(ang, &sn, &cc);
        o1 = x1 * cc - x2 * sn;
        o2 = x2 * cc + x1 * sn;
    }
    bf16* dp = dst + ((size_t)(b * nh + hh) * cS + s) * 64;
    dp[d]      = (bf16)o1;
    dp[d + 32] = (bf16)o2;
}

// ---------------------------------------------------------------------------
// 4b. V repack (transposed): src fp32 [B*S, HKV*64] -> dst bf16 [B, HKV, 64, S]
// ---------------------------------------------------------------------------
__global__ __launch_bounds__(256) void vpack_kernel(const float* __restrict__ src,
                                                    bf16* __restrict__ dst) {
    int t = blockIdx.x * blockDim.x + threadIdx.x;
    int total = cB * cHKV * cS * cHD;   // 524288
    if (t >= total) return;
    int d  = t & 63;
    int s  = (t >> 6) & (cS - 1);
    int gh = (t >> 16) & 3;
    int b  = t >> 18;
    float x = src[(size_t)(b * cS + s) * (cHKV * cHD) + gh * 64 + d];
    dst[((size_t)(b * cHKV + gh) * 64 + d) * cS + s] = (bf16)x;
}

// ---------------------------------------------------------------------------
// 5. Flash attention (causal, GQA). One wave per (b, h, 16-row q-tile).
//    KV tiles of 32. Scores via 4 WMMAs, online softmax with shfl row
//    reductions, P relayout C->A via per-wave LDS strip, AV via 4 WMMAs.
//    q: [B,H,S,64] bf16   k: [B,HKV,S,64] bf16   v: [B,HKV,64,S] bf16
//    out: [B*S, H*64] bf16
// ---------------------------------------------------------------------------
__global__ __launch_bounds__(256) void attn_kernel(const bf16* __restrict__ qr,
                                                   const bf16* __restrict__ kr,
                                                   const bf16* __restrict__ vt,
                                                   bf16* __restrict__ out) {
    __shared__ bf16 sP[8 * 16 * 32];    // one 16x32 strip per wave

    int tid   = threadIdx.x;
    int lane  = tid & 31;
    int wv    = tid >> 5;
    int w     = blockIdx.x * 8 + wv;    // 2048 waves total
    int mtile = w & 63;
    int bh    = w >> 6;
    int h     = bh & 15;
    int b     = bh >> 4;
    int g     = h >> 2;                 // kv head = h / NREP
    int m_base = mtile * 16;
    int lrow  = lane & 15;
    int half  = lane >> 4;
    bf16* pbase = sP + wv * (16 * 32);

    // Q fragments (A layout): row = m_base + lrow, two K-frags over d=0..63
    const bf16* qrow = qr + ((size_t)(b * cH + h) * cS + (m_base + lrow)) * 64;
    v16bf qf[2];
#pragma unroll
    for (int kf = 0; kf < 2; ++kf)
        qf[kf] = cat8(*(const v8bf*)(qrow + kf * 32 + half * 8),
                      *(const v8bf*)(qrow + kf * 32 + 16 + half * 8));

    v8f zero8 = { 0.f, 0.f, 0.f, 0.f, 0.f, 0.f, 0.f, 0.f };
    v8f acc[4];
#pragma unroll
    for (int i = 0; i < 4; ++i) acc[i] = zero8;
    float mrow[8], lsum[8];
#pragma unroll
    for (int r = 0; r < 8; ++r) { mrow[r] = -1e30f; lsum[r] = 0.f; }

    const float scale = 0.125f;   // 1/sqrt(64)
    const bf16* kbase = kr + (size_t)(b * cHKV + g) * cS * 64;
    const bf16* vbase = vt + (size_t)(b * cHKV + g) * 64 * cS;

    int jt_max = (m_base + 15) >> 5;    // wave-uniform loop bound (EXEC stays full)
    for (int jt = 0; jt <= jt_max; ++jt) {
        int jb = jt * 32;
        // ---- scores: S(16x32) = Q(16x64) @ K^T  as two 16x16 C-frags
        v8f sc0 = zero8, sc1 = zero8;
#pragma unroll
        for (int kf = 0; kf < 2; ++kf) {
            const bf16* kp0 = kbase + (size_t)(jb + lrow) * 64 + kf * 32 + half * 8;
            const bf16* kp1 = kbase + (size_t)(jb + 16 + lrow) * 64 + kf * 32 + half * 8;
            v16bf b0 = cat8(*(const v8bf*)kp0, *(const v8bf*)(kp0 + 16));
            v16bf b1 = cat8(*(const v8bf*)kp1, *(const v8bf*)(kp1 + 16));
            sc0 = wmma_bf16(qf[kf], b0, sc0);
            sc1 = wmma_bf16(qf[kf], b1, sc1);
        }
        // ---- online softmax (rows live in 16-lane halves; masks 1..8 stay in-half)
#pragma unroll
        for (int r = 0; r < 8; ++r) {
            int row = half ? r + 8 : r;
            int i   = m_base + row;
            float s0 = sc0[r] * scale;
            float s1 = sc1[r] * scale;
            if (jb + lrow > i)      s0 = -1e30f;   // causal mask (select, no branch)
            if (jb + 16 + lrow > i) s1 = -1e30f;
            float vmax = fmaxf(s0, s1);
#pragma unroll
            for (int m = 8; m >= 1; m >>= 1) vmax = fmaxf(vmax, __shfl_xor(vmax, m));
            float mnew = fmaxf(mrow[r], vmax);
            float corr = __expf(mrow[r] - mnew);
            float p0 = __expf(s0 - mnew);
            float p1 = __expf(s1 - mnew);
            float ps = p0 + p1;
#pragma unroll
            for (int m = 8; m >= 1; m >>= 1) ps += __shfl_xor(ps, m);
            lsum[r] = lsum[r] * corr + ps;
            mrow[r] = mnew;
#pragma unroll
            for (int vd = 0; vd < 4; ++vd) acc[vd][r] *= corr;
            pbase[row * 32 + lrow]      = (bf16)p0;
            pbase[row * 32 + 16 + lrow] = (bf16)p1;
        }
        // cross-lane LDS dependency: wait for this wave's P stores to land
        asm volatile("s_wait_dscnt 0" ::: "memory");
        // ---- P relayout: load as A fragment (16x32 bf16)
        const bf16* pr = pbase + lrow * 32;
        v16bf pa = cat8(*(const v8bf*)(pr + half * 8),
                        *(const v8bf*)(pr + 16 + half * 8));
        // ---- O += P @ V   (V transposed: contiguous along s)
#pragma unroll
        for (int vd = 0; vd < 4; ++vd) {
            const bf16* vp = vbase + (size_t)(vd * 16 + lrow) * cS + jb + half * 8;
            v16bf vfrag = cat8(*(const v8bf*)vp, *(const v8bf*)(vp + 16));
            acc[vd] = wmma_bf16(pa, vfrag, acc[vd]);
        }
    }

    // ---- epilogue: normalize and store bf16 [B*S, H*64]
    float rinv[8];
#pragma unroll
    for (int r = 0; r < 8; ++r) rinv[r] = 1.0f / lsum[r];
#pragma unroll
    for (int vd = 0; vd < 4; ++vd) {
#pragma unroll
        for (int r = 0; r < 8; ++r) {
            int row = half ? r + 8 : r;
            size_t idx = ((size_t)(b * cS) + m_base + row) * (cH * cHD)
                         + h * 64 + vd * 16 + lrow;
            out[idx] = (bf16)(acc[vd][r] * rinv[r]);
        }
    }
}

// ---------------------------------------------------------------------------
// 6. SiLU(gate) * up  ->  bf16
// ---------------------------------------------------------------------------
__global__ __launch_bounds__(256) void silu_mul_kernel(const float* __restrict__ g,
                                                       const float* __restrict__ u,
                                                       bf16* __restrict__ out, int n) {
    int t = blockIdx.x * blockDim.x + threadIdx.x;
    if (t >= n) return;
    float gv = g[t];
    float s  = gv / (1.0f + __expf(-gv));
    out[t] = (bf16)(s * u[t]);
}

// ---------------------------------------------------------------------------
// Host orchestration
// ---------------------------------------------------------------------------
extern "C" void kernel_launch(void* const* d_in, const int* in_sizes, int n_in,
                              void* d_out, int out_size, void* d_ws, size_t ws_size,
                              hipStream_t stream) {
    const int*   tokens   = (const int*)d_in[0];
    const float* embed    = (const float*)d_in[1];
    const float* wq       = (const float*)d_in[2];
    const float* wk       = (const float*)d_in[3];
    const float* wv       = (const float*)d_in[4];
    const float* wo       = (const float*)d_in[5];
    const float* w_gate   = (const float*)d_in[6];
    const float* w_up     = (const float*)d_in[7];
    const float* w_down   = (const float*)d_in[8];
    const float* norm1_w  = (const float*)d_in[9];
    const float* norm2_w  = (const float*)d_in[10];
    const float* final_w  = (const float*)d_in[11];
    float* logits = (float*)d_out;

    // ---- workspace carve-out (~120 MB)
    char* ws = (char*)d_ws;
    size_t off = 0;
    auto carve = [&](size_t bytes) -> void* {
        void* p = ws + off;
        off += (bytes + 255) & ~(size_t)255;
        return p;
    };
    float* x    = (float*)carve((size_t)cM * cD * 4);                 // residual
    bf16*  h    = (bf16*) carve((size_t)cM * cD * 2);                 // normed acts
    float* qb   = (float*)carve((size_t)cM * cH * cHD * 4);
    float* kb   = (float*)carve((size_t)cM * cHKV * cHD * 4);
    float* vb   = (float*)carve((size_t)cM * cHKV * cHD * 4);
    bf16*  q_r  = (bf16*) carve((size_t)cB * cH * cS * cHD * 2);
    bf16*  k_r  = (bf16*) carve((size_t)cB * cHKV * cS * cHD * 2);
    bf16*  v_t  = (bf16*) carve((size_t)cB * cHKV * cHD * cS * 2);
    bf16*  att  = (bf16*) carve((size_t)cM * cH * cHD * 2);
    float* gate = (float*)carve((size_t)cM * cFF * 4);
    float* up   = (float*)carve((size_t)cM * cFF * 4);
    bf16*  actb = (bf16*) carve((size_t)cM * cFF * 2);
    (void)ws_size; (void)in_sizes; (void)n_in; (void)out_size;

    dim3 blk(256);
    const int Mrows = cM;

    // embedding gather
    embed_kernel<<<Mrows, blk, 0, stream>>>(tokens, embed, x);

    for (int i = 0; i < cL; ++i) {
        const float* wq_i = wq + (size_t)i * cD * (cH * cHD);
        const float* wk_i = wk + (size_t)i * cD * (cHKV * cHD);
        const float* wv_i = wv + (size_t)i * cD * (cHKV * cHD);
        const float* wo_i = wo + (size_t)i * (cH * cHD) * cD;
        const float* wg_i = w_gate + (size_t)i * cD * cFF;
        const float* wu_i = w_up   + (size_t)i * cD * cFF;
        const float* wd_i = w_down + (size_t)i * cFF * cD;

        // h = rmsnorm(x, norm1) -> bf16
        rmsnorm_kernel<<<Mrows, blk, 0, stream>>>(x, norm1_w + (size_t)i * cD, h);

        // q/k/v projections
        gemm_kernel<false, false><<<dim3((cH * cHD) / 128, Mrows / 64), blk, 0, stream>>>(
            h, wq_i, qb, nullptr, Mrows, cH * cHD, cD);
        gemm_kernel<false, false><<<dim3((cHKV * cHD) / 128, Mrows / 64), blk, 0, stream>>>(
            h, wk_i, kb, nullptr, Mrows, cHKV * cHD, cD);
        gemm_kernel<false, false><<<dim3((cHKV * cHD) / 128, Mrows / 64), blk, 0, stream>>>(
            h, wv_i, vb, nullptr, Mrows, cHKV * cHD, cD);

        // RoPE + repack to attention layouts
        rope_pack_kernel<<<(cB * cH * cS * 32) / 256, blk, 0, stream>>>(qb, q_r, cH, 1);
        rope_pack_kernel<<<(cB * cHKV * cS * 32) / 256, blk, 0, stream>>>(kb, k_r, cHKV, 1);
        vpack_kernel<<<(cB * cHKV * cS * cHD) / 256, blk, 0, stream>>>(vb, v_t);

        // flash attention: 2048 waves / 8 per block = 256 blocks
        attn_kernel<<<256, blk, 0, stream>>>(q_r, k_r, v_t, att);

        // x += att @ wo
        gemm_kernel<false, true><<<dim3(cD / 128, Mrows / 64), blk, 0, stream>>>(
            att, wo_i, x, x, Mrows, cD, cH * cHD);

        // FFN
        rmsnorm_kernel<<<Mrows, blk, 0, stream>>>(x, norm2_w + (size_t)i * cD, h);
        gemm_kernel<false, false><<<dim3(cFF / 128, Mrows / 64), blk, 0, stream>>>(
            h, wg_i, gate, nullptr, Mrows, cFF, cD);
        gemm_kernel<false, false><<<dim3(cFF / 128, Mrows / 64), blk, 0, stream>>>(
            h, wu_i, up, nullptr, Mrows, cFF, cD);
        silu_mul_kernel<<<((size_t)Mrows * cFF) / 256, blk, 0, stream>>>(
            gate, up, actb, Mrows * cFF);
        gemm_kernel<false, true><<<dim3(cD / 128, Mrows / 64), blk, 0, stream>>>(
            actb, wd_i, x, x, Mrows, cD, cFF);
    }

    // final norm + logits = h @ embed^T  (B transposed: embed is [V, D])
    rmsnorm_kernel<<<Mrows, blk, 0, stream>>>(x, final_w, h);
    gemm_kernel<true, false><<<dim3(cV / 128, Mrows / 64), blk, 0, stream>>>(
        h, embed, logits, nullptr, Mrows, cV, cD);
}
